// AxialShift_37933151158895
// MI455X (gfx1250) — matrix operand order repaired and checked by reading, not compiled, not run on hardware
//
#include <hip/hip_runtime.h>
#include <cstdint>
#include <cstddef>

typedef _Float16 half_t;
typedef __attribute__((ext_vector_type(16))) _Float16 v16h;
typedef __attribute__((ext_vector_type(8)))  _Float16 v8h;
typedef __attribute__((ext_vector_type(8)))  float    v8f;

#define DIMC   384
#define NVOX   32768          // voxels per sample (32^3)
#define NTOT   65536          // B * NVOX
#define LDSROW 40             // padded halves per LDS tile row (32 data + 8 pad)
#define KTILES 12             // 384 / 32
#define WMAT   147456         // 384*384 elements per weight matrix
#define TILEH  (128 * LDSROW) // halves per LDS tile buffer

__device__ __forceinline__ float gelu_f(float x) {
  return 0.5f * x * (1.0f + erff(x * 0.7071067811865475f));
}

// ---- WMMA fragment loads from LDS --------------------------------------
// A (16x32 f16): lane half g: elems 0..7 -> K = 8g..8g+7 ; elems 8..15 -> K = 16+8g..
__device__ __forceinline__ v16h load_fragA(const half_t* base, int lq, int g) {
  const half_t* r = base + lq * LDSROW;
  v8h lo = *(const v8h*)(r + 8 * g);
  v8h hi = *(const v8h*)(r + 16 + 8 * g);
  v16h f;
#pragma unroll
  for (int e = 0; e < 8; ++e) { f[e] = lo[e]; f[8 + e] = hi[e]; }
  return f;
}
// B (32x16 f16): lane half g: elems 0..7 -> K = 8g..8g+7 ; elems 8..15 -> K = 16+8g.. (col = lane%16)
__device__ __forceinline__ v16h load_fragB(const half_t* base, int lq, int g) {
  const half_t* r = base + lq * LDSROW;
  v8h lo = *(const v8h*)(r + 8 * g);
  v8h hi = *(const v8h*)(r + 16 + 8 * g);
  v16h f;
#pragma unroll
  for (int e = 0; e < 8; ++e) { f[e] = lo[e]; f[8 + e] = hi[e]; }
  return f;
}

// 8 WMMAs: wave tile = 64 voxels (4 frags) x 32 out-channels (2 frags)
__device__ __forceinline__ void mma_tiles(const half_t* As, const half_t* Bs,
                                          int wm, int wn, int lq, int g,
                                          v8f (&acc)[4][2]) {
  v16h a[4], b[2];
#pragma unroll
  for (int i = 0; i < 4; ++i) a[i] = load_fragA(As + (wm * 64 + 16 * i) * LDSROW, lq, g);
#pragma unroll
  for (int j = 0; j < 2; ++j) b[j] = load_fragB(Bs + (wn * 32 + 16 * j) * LDSROW, lq, g);
#pragma unroll
  for (int i = 0; i < 4; ++i)
#pragma unroll
    for (int j = 0; j < 2; ++j)
      acc[i][j] = __builtin_amdgcn_wmma_f32_16x16x32_f16(
          false, a[i], false, b[j], (short)0, acc[i][j], false, false);
}

// ---- global -> register tile fetch (2 x 16B per thread) -----------------
__device__ __forceinline__ void fetch_plain(const half_t* base, int kk, int tid,
                                            uint4 (&r)[2]) {
#pragma unroll
  for (int it = 0; it < 2; ++it) {
    int c = tid + it * 256;
    int row = c >> 2, part = c & 3;
    r[it] = *(const uint4*)(base + (size_t)row * DIMC + kk + part * 8);
  }
}

// Shifted activation fetch for the axial-shift convs. sel: 0=d,1=h,2=w.
__device__ __forceinline__ void fetch_shift(const half_t* ha, int sample, int nLocalBase,
                                            int stride, int sel, int s, int kk, int tid,
                                            uint4 (&r)[2]) {
#pragma unroll
  for (int it = 0; it < 2; ++it) {
    int c = tid + it * 256;
    int row = c >> 2, part = c & 3;
    int nl = nLocalBase + row;
    int coord = (sel == 0) ? (nl >> 10) : ((sel == 1) ? ((nl >> 5) & 31) : (nl & 31));
    int c2 = coord - s;
    uint4 v; v.x = 0; v.y = 0; v.z = 0; v.w = 0;
    if ((unsigned)c2 < 32u) {
      int nsrc = nl - s * stride;
      v = *(const uint4*)(ha + (size_t)(sample * NVOX + nsrc) * DIMC + kk + part * 8);
    }
    r[it] = v;
  }
}

// register tile -> LDS (padded rows)
__device__ __forceinline__ void store_tile(half_t* dst, int tid, const uint4 (&r)[2]) {
#pragma unroll
  for (int it = 0; it < 2; ++it) {
    int c = tid + it * 256;
    int row = c >> 2, part = c & 3;
    *(uint4*)(dst + row * LDSROW + part * 8) = r[it];
  }
}

// ---- utility kernels ----------------------------------------------------
__global__ __launch_bounds__(256) void cvt_weights_kernel(
    const float* w1, const float* w21, const float* w22, const float* w23,
    const float* w3, half_t* dst) {
  int i = blockIdx.x * 256 + threadIdx.x;
  if (i < WMAT) {
    dst[i]             = (half_t)w1[i];
    dst[WMAT + i]      = (half_t)w21[i];
    dst[2 * WMAT + i]  = (half_t)w22[i];
    dst[3 * WMAT + i]  = (half_t)w23[i];
    dst[4 * WMAT + i]  = (half_t)w3[i];
  }
}

// x fp32 [b][c][vox] -> xT f16 [b*vox][c]
__global__ __launch_bounds__(256) void transpose_x_kernel(const float* x, half_t* xT) {
  __shared__ float t[32][33];
  int nBase = blockIdx.x * 32;            // global voxel
  int cBase = blockIdx.y * 32;
  int sample = nBase >> 15;
  int nLoc = nBase & (NVOX - 1);
  int tx = threadIdx.x & 31, ty = threadIdx.x >> 5;
#pragma unroll
  for (int yy = ty; yy < 32; yy += 8)
    t[yy][tx] = x[((size_t)(sample * DIMC + cBase + yy)) * NVOX + nLoc + tx];
  __syncthreads();
#pragma unroll
  for (int yy = ty; yy < 32; yy += 8)
    xT[((size_t)(nBase + yy)) * DIMC + cBase + tx] = (half_t)t[tx][yy];
}

// Deterministic fixed-order reduction of per-block partials -> mean/rstd per sample.
__global__ void reduce_stats_kernel(const float* partials, float* stats) {
  int s = blockIdx.x;
  if (threadIdx.x != 0) return;
  float sum = 0.f, sq = 0.f;
  for (int y = 0; y < 3; ++y)
    for (int xx = 0; xx < 256; ++xx) {
      int idx = y * 512 + s * 256 + xx;
      sum += partials[2 * idx];
      sq  += partials[2 * idx + 1];
    }
  float cnt = (float)DIMC * (float)NVOX;
  float mean = sum / cnt;
  float var = sq / cnt - mean * mean;
  stats[s]     = mean;
  stats[2 + s] = rsqrtf(var + 1e-5f);
}

// In-place: h = gelu((h-mean)*rstd*g + b)   (stage 1 epilogue)
__global__ __launch_bounds__(256) void finalize1_kernel(half_t* h, const float* stats,
                                                        const float* g1, const float* bt1) {
  size_t base = ((size_t)blockIdx.x * 256 + threadIdx.x) * 8;
  int sample = (base >= (size_t)DIMC * NVOX) ? 1 : 0;
  int c0 = (int)(base % DIMC);
  float mean = stats[sample], rstd = stats[2 + sample];
  uint4 raw = *(uint4*)(h + base);
  half_t* hp = (half_t*)&raw;
#pragma unroll
  for (int e = 0; e < 8; ++e) {
    int c = c0 + e;
    float v = (float)hp[e];
    v = (v - mean) * rstd * g1[c] + bt1[c];
    hp[e] = (half_t)gelu_f(v);
  }
  *(uint4*)(h + base) = raw;
}

// In-place: y = (y-mean)*rstd*g + b   (stage 2 epilogue, no gelu)
__global__ __launch_bounds__(256) void finalize2_kernel(half_t* y, const float* stats,
                                                        const float* g2, const float* bt2) {
  size_t base = ((size_t)blockIdx.x * 256 + threadIdx.x) * 8;
  int sample = (base >= (size_t)DIMC * NVOX) ? 1 : 0;
  int c0 = (int)(base % DIMC);
  float mean = stats[sample], rstd = stats[2 + sample];
  uint4 raw = *(uint4*)(y + base);
  half_t* hp = (half_t*)&raw;
#pragma unroll
  for (int e = 0; e < 8; ++e) {
    int c = c0 + e;
    float v = (float)hp[e];
    v = (v - mean) * rstd * g2[c] + bt2[c];
    hp[e] = (half_t)v;
  }
  *(uint4*)(y + base) = raw;
}

// ---- stage 1: h0 = x @ w1^T + b1 ; write f16 [vox][o] + stats partials --
// Register-prefetch + LDS ping-pong pipeline: 1 barrier per K-tile.
__global__ __launch_bounds__(256) void gemm1_kernel(const half_t* xT, const half_t* w1h,
                                                    const float* b1, half_t* h0,
                                                    float* partials) {
  __shared__ half_t AsB[2][TILEH];
  __shared__ half_t BsB[2][TILEH];
  __shared__ float rs[256], rq[256];
  int tid = threadIdx.x, wave = tid >> 5, lane = tid & 31;
  int lq = lane & 15, g = (lane >> 4) & 1;
  int wm = wave >> 2, wn = wave & 3;
  int nBaseG = blockIdx.x * 128;
  int oBase0 = blockIdx.y * 128;
  const half_t* aBase = xT + (size_t)nBaseG * DIMC;
  const half_t* bBase = w1h + (size_t)oBase0 * DIMC;

  v8f acc[4][2];
#pragma unroll
  for (int i = 0; i < 4; ++i)
#pragma unroll
    for (int j = 0; j < 2; ++j) acc[i][j] = (v8f)(0.0f);

  uint4 ra[2], rb[2];
  fetch_plain(aBase, 0, tid, ra);
  fetch_plain(bBase, 0, tid, rb);
  store_tile(AsB[0], tid, ra);
  store_tile(BsB[0], tid, rb);
  __syncthreads();

  for (int kt = 0; kt < KTILES; ++kt) {
    int cur = kt & 1;
    if (kt < KTILES - 1) {
      fetch_plain(aBase, (kt + 1) * 32, tid, ra);   // overlap with WMMAs below
      fetch_plain(bBase, (kt + 1) * 32, tid, rb);
    }
    mma_tiles(AsB[cur], BsB[cur], wm, wn, lq, g, acc);
    if (kt < KTILES - 1) {
      store_tile(AsB[cur ^ 1], tid, ra);
      store_tile(BsB[cur ^ 1], tid, rb);
    }
    __syncthreads();
  }

  float lsum = 0.f, lsq = 0.f;
#pragma unroll
  for (int j = 0; j < 2; ++j) {
    int o = oBase0 + wn * 32 + 16 * j + lq;
    float bias = b1[o];
#pragma unroll
    for (int i = 0; i < 4; ++i)
#pragma unroll
      for (int r = 0; r < 8; ++r) {
        int n = nBaseG + wm * 64 + 16 * i + r + 8 * g;
        float v = acc[i][j][r] + bias;
        lsum += v; lsq += v * v;
        h0[(size_t)n * DIMC + o] = (half_t)v;
      }
  }
  rs[tid] = lsum; rq[tid] = lsq;
  __syncthreads();
  for (int s = 128; s > 0; s >>= 1) {
    if (tid < s) { rs[tid] += rs[tid + s]; rq[tid] += rq[tid + s]; }
    __syncthreads();
  }
  if (tid == 0) {
    int idx = blockIdx.y * 512 + blockIdx.x;
    partials[2 * idx] = rs[0]; partials[2 * idx + 1] = rq[0];
  }
}

// ---- stage 2: y = sum_axes gelu(conv(shift_axis(ha), w2a) + b2a) --------
__global__ __launch_bounds__(256) void gemm2_kernel(const half_t* ha, const half_t* w2h,
                                                    const float* b21, const float* b22,
                                                    const float* b23, half_t* y,
                                                    float* partials) {
  __shared__ half_t AsB[2][TILEH];
  __shared__ half_t BsB[2][TILEH];
  __shared__ float rs[256], rq[256];
  int tid = threadIdx.x, wave = tid >> 5, lane = tid & 31;
  int lq = lane & 15, g = (lane >> 4) & 1;
  int wm = wave >> 2, wn = wave & 3;
  int nBaseG = blockIdx.x * 128;
  int oBase0 = blockIdx.y * 128;
  int sample = nBaseG >> 15;
  int nLocalBase = nBaseG & (NVOX - 1);

  v8f accS[4][2];
#pragma unroll
  for (int i = 0; i < 4; ++i)
#pragma unroll
    for (int j = 0; j < 2; ++j) accS[i][j] = (v8f)(0.0f);

#pragma unroll
  for (int a = 0; a < 3; ++a) {
    // a=0: x_lr (h-shift), a=1: x_td (d-shift), a=2: x_hd (w-shift)
    int sel    = (a == 0) ? 1 : ((a == 1) ? 0 : 2);
    int stride = (a == 0) ? 32 : ((a == 1) ? 1024 : 1);
    const half_t* wb = w2h + (size_t)a * WMAT + (size_t)oBase0 * DIMC;
    const float* bp = (a == 0) ? b21 : ((a == 1) ? b22 : b23);

    v8f acc[4][2];
#pragma unroll
    for (int i = 0; i < 4; ++i)
#pragma unroll
      for (int j = 0; j < 2; ++j) acc[i][j] = (v8f)(0.0f);

    uint4 ra[2], rb[2];
    fetch_shift(ha, sample, nLocalBase, stride, sel, -1, 0, tid, ra);
    fetch_plain(wb, 0, tid, rb);
    store_tile(AsB[0], tid, ra);
    store_tile(BsB[0], tid, rb);
    __syncthreads();

    for (int kt = 0; kt < KTILES; ++kt) {
      int cur = kt & 1;
      if (kt < KTILES - 1) {
        int kk2 = (kt + 1) * 32;
        int s2 = (kk2 >> 7) - 1;            // channel-chunk shift: -1,0,+1
        fetch_shift(ha, sample, nLocalBase, stride, sel, s2, kk2, tid, ra);
        fetch_plain(wb, kk2, tid, rb);
      }
      mma_tiles(AsB[cur], BsB[cur], wm, wn, lq, g, acc);
      if (kt < KTILES - 1) {
        store_tile(AsB[cur ^ 1], tid, ra);
        store_tile(BsB[cur ^ 1], tid, rb);
      }
      __syncthreads();
    }

#pragma unroll
    for (int j = 0; j < 2; ++j) {
      int o = oBase0 + wn * 32 + 16 * j + lq;
      float bias = bp[o];
#pragma unroll
      for (int i = 0; i < 4; ++i)
#pragma unroll
        for (int r = 0; r < 8; ++r)
          accS[i][j][r] += gelu_f(acc[i][j][r] + bias);
    }
  }

  float lsum = 0.f, lsq = 0.f;
#pragma unroll
  for (int j = 0; j < 2; ++j) {
    int o = oBase0 + wn * 32 + 16 * j + lq;
#pragma unroll
    for (int i = 0; i < 4; ++i)
#pragma unroll
      for (int r = 0; r < 8; ++r) {
        int n = nBaseG + wm * 64 + 16 * i + r + 8 * g;
        float v = accS[i][j][r];
        lsum += v; lsq += v * v;
        y[(size_t)n * DIMC + o] = (half_t)v;
      }
  }
  rs[tid] = lsum; rq[tid] = lsq;
  __syncthreads();
  for (int s = 128; s > 0; s >>= 1) {
    if (tid < s) { rs[tid] += rs[tid + s]; rq[tid] += rq[tid + s]; }
    __syncthreads();
  }
  if (tid == 0) {
    int idx = blockIdx.y * 512 + blockIdx.x;
    partials[2 * idx] = rs[0]; partials[2 * idx + 1] = rq[0];
  }
}

// ---- stage 3: out = yn @ w3^T + b3, fp32 output in [b][o][vox] layout ---
__global__ __launch_bounds__(256) void gemm3_kernel(const half_t* yn, const half_t* w3h,
                                                    const float* b3, float* out) {
  __shared__ half_t AsB[2][TILEH];
  __shared__ half_t BsB[2][TILEH];
  __shared__ float ltr[128 * 140];
  int tid = threadIdx.x, wave = tid >> 5, lane = tid & 31;
  int lq = lane & 15, g = (lane >> 4) & 1;
  int wm = wave >> 2, wn = wave & 3;
  int nBaseG = blockIdx.x * 128;
  int oBase0 = blockIdx.y * 128;
  const half_t* aBase = yn + (size_t)nBaseG * DIMC;
  const half_t* bBase = w3h + (size_t)oBase0 * DIMC;

  v8f acc[4][2];
#pragma unroll
  for (int i = 0; i < 4; ++i)
#pragma unroll
    for (int j = 0; j < 2; ++j) acc[i][j] = (v8f)(0.0f);

  uint4 ra[2], rb[2];
  fetch_plain(aBase, 0, tid, ra);
  fetch_plain(bBase, 0, tid, rb);
  store_tile(AsB[0], tid, ra);
  store_tile(BsB[0], tid, rb);
  __syncthreads();

  for (int kt = 0; kt < KTILES; ++kt) {
    int cur = kt & 1;
    if (kt < KTILES - 1) {
      fetch_plain(aBase, (kt + 1) * 32, tid, ra);
      fetch_plain(bBase, (kt + 1) * 32, tid, rb);
    }
    mma_tiles(AsB[cur], BsB[cur], wm, wn, lq, g, acc);
    if (kt < KTILES - 1) {
      store_tile(AsB[cur ^ 1], tid, ra);
      store_tile(BsB[cur ^ 1], tid, rb);
    }
    __syncthreads();
  }

  // transpose through LDS: ltr[o_local][vox_local]
#pragma unroll
  for (int j = 0; j < 2; ++j) {
    int ol = wn * 32 + 16 * j + lq;
    float bias = b3[oBase0 + ol];
#pragma unroll
    for (int i = 0; i < 4; ++i)
#pragma unroll
      for (int r = 0; r < 8; ++r) {
        int vl = wm * 64 + 16 * i + r + 8 * g;
        ltr[ol * 140 + vl] = acc[i][j][r] + bias;
      }
  }
  __syncthreads();
  int sample = nBaseG >> 15;
  int nLoc = nBaseG & (NVOX - 1);
  int row = tid >> 1, half = tid & 1;
  const float* srcp = ltr + row * 140 + half * 64;
  float* dstp = out + ((size_t)(sample * DIMC + oBase0 + row)) * NVOX + nLoc + half * 64;
#pragma unroll
  for (int q = 0; q < 16; ++q)
    ((float4*)dstp)[q] = ((const float4*)srcp)[q];
}

// ---- host side ----------------------------------------------------------
extern "C" void kernel_launch(void* const* d_in, const int* in_sizes, int n_in,
                              void* d_out, int out_size, void* d_ws, size_t ws_size,
                              hipStream_t stream) {
  const float* x   = (const float*)d_in[0];
  const float* w1  = (const float*)d_in[1];
  const float* b1  = (const float*)d_in[2];
  const float* g1  = (const float*)d_in[3];
  const float* bt1 = (const float*)d_in[4];
  const float* w21 = (const float*)d_in[5];
  const float* b21 = (const float*)d_in[6];
  const float* w22 = (const float*)d_in[7];
  const float* b22 = (const float*)d_in[8];
  const float* w23 = (const float*)d_in[9];
  const float* b23 = (const float*)d_in[10];
  const float* g2  = (const float*)d_in[11];
  const float* bt2 = (const float*)d_in[12];
  const float* w3  = (const float*)d_in[13];
  const float* b3  = (const float*)d_in[14];
  float* out = (float*)d_out;

  char* ws = (char*)d_ws;
  const size_t SZ_ACT = (size_t)NTOT * DIMC * sizeof(half_t);  // 50,331,648
  half_t* xT = (half_t*)(ws);
  half_t* ha = (half_t*)(ws + SZ_ACT);
  half_t* yb = (half_t*)(ws + 2 * SZ_ACT);
  half_t* wh = (half_t*)(ws + 3 * SZ_ACT);
  float* part1  = (float*)(ws + 3 * SZ_ACT + 5 * WMAT * sizeof(half_t));
  float* part2  = part1 + 3072;
  float* stats1 = part2 + 3072;
  float* stats2 = stats1 + 4;

  // 0) weights -> f16
  cvt_weights_kernel<<<(WMAT + 255) / 256, 256, 0, stream>>>(w1, w21, w22, w23, w3, wh);
  // 1) x -> xT f16 [vox][c]
  transpose_x_kernel<<<dim3(NTOT / 32, DIMC / 32), 256, 0, stream>>>(x, xT);
  // 2) GEMM1 + stats partials
  gemm1_kernel<<<dim3(NTOT / 128, DIMC / 128), 256, 0, stream>>>(xT, wh, b1, ha, part1);
  // 3) stats1
  reduce_stats_kernel<<<2, 32, 0, stream>>>(part1, stats1);
  // 4) norm1 + gelu in place
  finalize1_kernel<<<(NTOT * DIMC / 8) / 256, 256, 0, stream>>>(ha, stats1, g1, bt1);
  // 5) fused shifted convs + gelu + sum -> y f16, partials
  gemm2_kernel<<<dim3(NTOT / 128, DIMC / 128), 256, 0, stream>>>(ha, wh + WMAT, b21, b22,
                                                                 b23, yb, part2);
  // 6) stats2
  reduce_stats_kernel<<<2, 32, 0, stream>>>(part2, stats2);
  // 7) norm2 in place
  finalize2_kernel<<<(NTOT * DIMC / 8) / 256, 256, 0, stream>>>(yb, stats2, g2, bt2);
  // 8) final GEMM -> fp32 output in reference layout
  gemm3_kernel<<<dim3(NTOT / 128, DIMC / 128), 256, 0, stream>>>(yb, wh + 4 * WMAT, b3, out);
}
